// GNSLayer_30494267802176
// MI455X (gfx1250) — compile-verified, matrix operand order checked by reading, not run on hardware
//
#include <hip/hip_runtime.h>

// ---------------------------------------------------------------------------
// GNS layer for MI455X (gfx1250, wave32).
// Edge MLP + LN + residual + scatter-add, then Node MLP + LN + residual.
// GEMMs via v_wmma_f32_16x16x32_bf16 (f32 accumulate). Weights staged in the
// 320KB WGP LDS once per block (bank-conflict padded); B fragments batched
// into registers ahead of each 8-WMMA chain so loads overlap the matrix pipe.
// ---------------------------------------------------------------------------

#define D_DIM   128
#define E_TOT   500000
#define N_TOT   50000
#define LN_EPS  1e-5f

// LDS row strides (elements), padded so lane-to-lane row stride rotates banks:
// stride_dw mod 64 = 4  ->  lane n starts at bank 4n (conflict-light b128 reads)
#define W1E_STRIDE 392   // 384+8  (edge layer-1 weights, K=384)
#define WN1_STRIDE 264   // 256+8  (node layer-1 weights, K=256)
#define W2_STRIDE  136   // 128+8  (layer-2 weights, K=128)
#define HT_STRIDE  136   // h/g tile rows, K=128

typedef __bf16 bf16_t;
typedef __attribute__((ext_vector_type(16))) bf16_t v16bf;
typedef __attribute__((ext_vector_type(8)))  bf16_t v8bf;
typedef __attribute__((ext_vector_type(8)))  float  v8f;

__device__ __forceinline__ v8f wmma_bf16(v16bf a, v16bf b, v8f c) {
    return __builtin_amdgcn_wmma_f32_16x16x32_bf16(
        false, a, false, b, (short)0, c, false, false);
}

// Fragment layout for one 32-K chunk (wave32):
//   lanes 0-15  (g=0): K = {0..7} U {16..23}
//   lanes 16-31 (g=1): K = {8..15} U {24..31}
__device__ __forceinline__ v16bf a_frag_from_f32(const float* __restrict__ chunk, int g) {
    const float4* pa = (const float4*)(chunk + 8 * g);
    const float4* pb = (const float4*)(chunk + 16 + 8 * g);
    float4 x0 = pa[0], x1 = pa[1];
    float4 y0 = pb[0], y1 = pb[1];
    v16bf r;
    r[0]  = (bf16_t)x0.x; r[1]  = (bf16_t)x0.y; r[2]  = (bf16_t)x0.z; r[3]  = (bf16_t)x0.w;
    r[4]  = (bf16_t)x1.x; r[5]  = (bf16_t)x1.y; r[6]  = (bf16_t)x1.z; r[7]  = (bf16_t)x1.w;
    r[8]  = (bf16_t)y0.x; r[9]  = (bf16_t)y0.y; r[10] = (bf16_t)y0.z; r[11] = (bf16_t)y0.w;
    r[12] = (bf16_t)y1.x; r[13] = (bf16_t)y1.y; r[14] = (bf16_t)y1.z; r[15] = (bf16_t)y1.w;
    return r;
}

__device__ __forceinline__ v16bf frag_from_bf16(const bf16_t* chunk, int g) {
    v8bf a = *(const v8bf*)(chunk + 8 * g);
    v8bf b = *(const v8bf*)(chunk + 16 + 8 * g);
    v16bf r;
#pragma unroll
    for (int i = 0; i < 8; ++i) { r[i] = a[i]; r[8 + i] = b[i]; }
    return r;
}

// ---------------------------------------------------------------------------
// Weight prep: transpose fp32 [K,N] -> bf16 [N,K] (dense rows, no pad) so LDS
// staging is straight 16B copies. 224 KB total -> L2 resident.
// ---------------------------------------------------------------------------
__global__ void prep_weights(const float* __restrict__ We1, const float* __restrict__ We2,
                             const float* __restrict__ Wn1, const float* __restrict__ Wn2,
                             bf16_t* __restrict__ we1t, bf16_t* __restrict__ we2t,
                             bf16_t* __restrict__ wn1t, bf16_t* __restrict__ wn2t) {
    int i = blockIdx.x * blockDim.x + threadIdx.x;
    if (i < 384 * 128) {                     // We1T [128][384]
        int n = i / 384, k = i % 384;
        we1t[i] = (bf16_t)We1[k * 128 + n];
        return;
    }
    int j = i - 384 * 128;
    if (j < 128 * 128) {                     // We2T [128][128]
        int n = j / 128, k = j % 128;
        we2t[j] = (bf16_t)We2[k * 128 + n];
        return;
    }
    int p = j - 128 * 128;
    if (p < 128 * 256) {                     // Wn1T [128][256]
        int n = p / 256, k = p % 256;
        wn1t[p] = (bf16_t)Wn1[k * 128 + n];
        return;
    }
    int q = p - 128 * 256;
    if (q < 128 * 128) {                     // Wn2T [128][128]
        int n = q / 128, k = q % 128;
        wn2t[q] = (bf16_t)Wn2[k * 128 + n];
    }
}

// Copy a dense bf16 [128][rowlen] matrix from global into LDS with padded rows.
__device__ __forceinline__ void stage_weights(const bf16_t* __restrict__ src,
                                              bf16_t* dst, int rowlen, int dstride,
                                              int tid, int nthreads) {
    const int chunks_per_row = rowlen / 8;          // 16B chunks
    const int total = 128 * chunks_per_row;
    const uint4* s = (const uint4*)src;
    for (int t = tid; t < total; t += nthreads) {
        int row = t / chunks_per_row, c = t % chunks_per_row;
        *(uint4*)(dst + row * dstride + c * 8) = s[t];
    }
}

// ---------------------------------------------------------------------------
// Edge kernel: 8 waves/block, 16 edges/wave (128 edges/block).
// ---------------------------------------------------------------------------
__global__ __launch_bounds__(256) void edge_kernel(
    const float* __restrict__ x, const float* __restrict__ ea,
    const float* __restrict__ be1, const float* __restrict__ be2,
    const float* __restrict__ gamma, const float* __restrict__ beta,
    const long long* __restrict__ eidx,
    const bf16_t* __restrict__ W1T,   // [128][384] dense
    const bf16_t* __restrict__ W2T,   // [128][128] dense
    float* __restrict__ out_edge, float* __restrict__ node_agg) {

    __shared__ __align__(16) bf16_t W1s[128 * W1E_STRIDE];      // 100352 B
    __shared__ __align__(16) bf16_t W2s[128 * W2_STRIDE];       //  34816 B
    __shared__ __align__(16) char   htile[8 * 16 * HT_STRIDE * 4]; // 69632 B (bf16 h1 / f32 g alias)

    const int w    = threadIdx.x >> 5;
    const int lane = threadIdx.x & 31;
    const int g    = lane >> 4;
    const int ln   = lane & 15;
    const int e0   = blockIdx.x * 128 + w * 16;

    stage_weights(W1T, W1s, 384, W1E_STRIDE, threadIdx.x, 256);
    stage_weights(W2T, W2s, 128, W2_STRIDE,  threadIdx.x, 256);

    int eA = e0 + ln; if (eA >= E_TOT) eA = E_TOT - 1;
    const int snd = (int)eidx[eA];
    const int rcv = (int)eidx[E_TOT + eA];

    v8f acc[8];
#pragma unroll
    for (int j = 0; j < 8; ++j) {
        float bv = be1[16 * j + ln];
#pragma unroll
        for (int v = 0; v < 8; ++v) acc[j][v] = bv;
    }
    __syncthreads();   // weights staged

    // ---- Layer 1: [16,384] @ [384,128] ----
#pragma unroll
    for (int c = 0; c < 12; ++c) {
        const float* src;
        if (c < 4)      src = x  + (size_t)snd * D_DIM + 32 * c;
        else if (c < 8) src = x  + (size_t)rcv * D_DIM + 32 * (c - 4);
        else            src = ea + (size_t)eA  * D_DIM + 32 * (c - 8);
        v16bf af = a_frag_from_f32(src, g);
        v16bf bfr[8];
#pragma unroll
        for (int j = 0; j < 8; ++j)
            bfr[j] = frag_from_bf16(W1s + (16 * j + ln) * W1E_STRIDE + 32 * c, g);
#pragma unroll
        for (int j = 0; j < 8; ++j)
            acc[j] = wmma_bf16(af, bfr[j], acc[j]);
    }

    // ---- ReLU -> LDS bf16 tile ----
    bf16_t* hb = (bf16_t*)(htile + w * 16 * HT_STRIDE * 4);
#pragma unroll
    for (int j = 0; j < 8; ++j)
#pragma unroll
        for (int v = 0; v < 8; ++v) {
            float t = acc[j][v]; t = t > 0.f ? t : 0.f;
            hb[(v + 8 * g) * HT_STRIDE + 16 * j + ln] = (bf16_t)t;
        }
    __syncthreads();

    // ---- Layer 2: [16,128] @ [128,128] ----
    v8f acc2[8];
#pragma unroll
    for (int j = 0; j < 8; ++j) {
        float bv = be2[16 * j + ln];
#pragma unroll
        for (int v = 0; v < 8; ++v) acc2[j][v] = bv;
    }
#pragma unroll
    for (int c = 0; c < 4; ++c) {
        v16bf af = frag_from_bf16(hb + ln * HT_STRIDE + 32 * c, g);
        v16bf bfr[8];
#pragma unroll
        for (int j = 0; j < 8; ++j)
            bfr[j] = frag_from_bf16(W2s + (16 * j + ln) * W2_STRIDE + 32 * c, g);
#pragma unroll
        for (int j = 0; j < 8; ++j)
            acc2[j] = wmma_bf16(af, bfr[j], acc2[j]);
    }

    // ---- stash g (f32) in LDS (aliases h1; same-wave DS ops stay in-order) ----
    float* gb = (float*)(htile + w * 16 * HT_STRIDE * 4);
#pragma unroll
    for (int j = 0; j < 8; ++j)
#pragma unroll
        for (int v = 0; v < 8; ++v)
            gb[(v + 8 * g) * HT_STRIDE + 16 * j + ln] = acc2[j][v];
    __syncthreads();

    // ---- LayerNorm + residual + scatter: lanes {ln, ln+16} share row ln ----
    const int em = e0 + ln;
    const float* grow = gb + ln * HT_STRIDE + g * 64;
    float s = 0.f, s2 = 0.f;
#pragma unroll
    for (int k = 0; k < 64; ++k) { float t = grow[k]; s += t; s2 += t * t; }
    s  += __shfl_xor(s, 16);
    s2 += __shfl_xor(s2, 16);
    const float mu  = s * (1.f / 128.f);
    const float var = s2 * (1.f / 128.f) - mu * mu;
    const float inv = rsqrtf(var + LN_EPS);

    if (em < E_TOT) {
        const int rv = (int)eidx[E_TOT + em];
        const float* earow  = ea       + (size_t)em * D_DIM + g * 64;
        float*       orow   = out_edge + (size_t)em * D_DIM + g * 64;
        float*       aggrow = node_agg + (size_t)rv * D_DIM + g * 64;
        const float* gm = gamma + g * 64;
        const float* bt = beta  + g * 64;
#pragma unroll
        for (int k = 0; k < 64; ++k) {
            float val = earow[k] + (grow[k] - mu) * inv * gm[k] + bt[k];
            orow[k] = val;
            atomicAdd(&aggrow[k], val);
        }
    }
}

// ---------------------------------------------------------------------------
// Node kernel: A = concat(x, node_agg)[16,256], 8 K-steps then layer 2.
// ---------------------------------------------------------------------------
__global__ __launch_bounds__(256) void node_kernel(
    const float* __restrict__ x, const float* __restrict__ agg,
    const float* __restrict__ bn1, const float* __restrict__ bn2,
    const float* __restrict__ gamma, const float* __restrict__ beta,
    const bf16_t* __restrict__ W1T,   // [128][256] dense
    const bf16_t* __restrict__ W2T,   // [128][128] dense
    float* __restrict__ out_x) {

    __shared__ __align__(16) bf16_t W1s[128 * WN1_STRIDE];      // 67584 B
    __shared__ __align__(16) bf16_t W2s[128 * W2_STRIDE];       // 34816 B
    __shared__ __align__(16) char   htile[8 * 16 * HT_STRIDE * 4]; // 69632 B

    const int w    = threadIdx.x >> 5;
    const int lane = threadIdx.x & 31;
    const int g    = lane >> 4;
    const int ln   = lane & 15;
    const int n0   = blockIdx.x * 128 + w * 16;

    stage_weights(W1T, W1s, 256, WN1_STRIDE, threadIdx.x, 256);
    stage_weights(W2T, W2s, 128, W2_STRIDE,  threadIdx.x, 256);

    int nA = n0 + ln; if (nA >= N_TOT) nA = N_TOT - 1;

    v8f acc[8];
#pragma unroll
    for (int j = 0; j < 8; ++j) {
        float bv = bn1[16 * j + ln];
#pragma unroll
        for (int v = 0; v < 8; ++v) acc[j][v] = bv;
    }
    __syncthreads();

#pragma unroll
    for (int c = 0; c < 8; ++c) {
        const float* src = (c < 4) ? (x   + (size_t)nA * D_DIM + 32 * c)
                                   : (agg + (size_t)nA * D_DIM + 32 * (c - 4));
        v16bf af = a_frag_from_f32(src, g);
        v16bf bfr[8];
#pragma unroll
        for (int j = 0; j < 8; ++j)
            bfr[j] = frag_from_bf16(W1s + (16 * j + ln) * WN1_STRIDE + 32 * c, g);
#pragma unroll
        for (int j = 0; j < 8; ++j)
            acc[j] = wmma_bf16(af, bfr[j], acc[j]);
    }

    bf16_t* hb = (bf16_t*)(htile + w * 16 * HT_STRIDE * 4);
#pragma unroll
    for (int j = 0; j < 8; ++j)
#pragma unroll
        for (int v = 0; v < 8; ++v) {
            float t = acc[j][v]; t = t > 0.f ? t : 0.f;
            hb[(v + 8 * g) * HT_STRIDE + 16 * j + ln] = (bf16_t)t;
        }
    __syncthreads();

    v8f acc2[8];
#pragma unroll
    for (int j = 0; j < 8; ++j) {
        float bv = bn2[16 * j + ln];
#pragma unroll
        for (int v = 0; v < 8; ++v) acc2[j][v] = bv;
    }
#pragma unroll
    for (int c = 0; c < 4; ++c) {
        v16bf af = frag_from_bf16(hb + ln * HT_STRIDE + 32 * c, g);
        v16bf bfr[8];
#pragma unroll
        for (int j = 0; j < 8; ++j)
            bfr[j] = frag_from_bf16(W2s + (16 * j + ln) * W2_STRIDE + 32 * c, g);
#pragma unroll
        for (int j = 0; j < 8; ++j)
            acc2[j] = wmma_bf16(af, bfr[j], acc2[j]);
    }

    float* gb = (float*)(htile + w * 16 * HT_STRIDE * 4);
#pragma unroll
    for (int j = 0; j < 8; ++j)
#pragma unroll
        for (int v = 0; v < 8; ++v)
            gb[(v + 8 * g) * HT_STRIDE + 16 * j + ln] = acc2[j][v];
    __syncthreads();

    const int nm = n0 + ln;
    const float* grow = gb + ln * HT_STRIDE + g * 64;
    float s = 0.f, s2 = 0.f;
#pragma unroll
    for (int k = 0; k < 64; ++k) { float t = grow[k]; s += t; s2 += t * t; }
    s  += __shfl_xor(s, 16);
    s2 += __shfl_xor(s2, 16);
    const float mu  = s * (1.f / 128.f);
    const float var = s2 * (1.f / 128.f) - mu * mu;
    const float inv = rsqrtf(var + LN_EPS);

    if (nm < N_TOT) {
        const float* xrow = x     + (size_t)nm * D_DIM + g * 64;
        float*       orow = out_x + (size_t)nm * D_DIM + g * 64;
        const float* gm = gamma + g * 64;
        const float* bt = beta  + g * 64;
#pragma unroll
        for (int k = 0; k < 64; ++k)
            orow[k] = xrow[k] + (grow[k] - mu) * inv * gm[k] + bt[k];
    }
}

// ---------------------------------------------------------------------------
// Launch. Workspace layout (bytes):
//   [0,      98304)  We1T bf16 [128][384]
//   [98304, 131072)  We2T bf16 [128][128]
//   [131072,196608)  Wn1T bf16 [128][256]
//   [196608,229376)  Wn2T bf16 [128][128]
//   [229376, +25.6M) node_agg f32 [50000][128]   (zeroed each call)
// ---------------------------------------------------------------------------
extern "C" void kernel_launch(void* const* d_in, const int* in_sizes, int n_in,
                              void* d_out, int out_size, void* d_ws, size_t ws_size,
                              hipStream_t stream) {
    const float*     x    = (const float*)d_in[0];
    const float*     ea   = (const float*)d_in[1];
    const float*     We1  = (const float*)d_in[2];
    const float*     be1  = (const float*)d_in[3];
    const float*     We2  = (const float*)d_in[4];
    const float*     be2  = (const float*)d_in[5];
    const float*     Wn1  = (const float*)d_in[6];
    const float*     bn1  = (const float*)d_in[7];
    const float*     Wn2  = (const float*)d_in[8];
    const float*     bn2  = (const float*)d_in[9];
    const float*     geg  = (const float*)d_in[10];
    const float*     geb  = (const float*)d_in[11];
    const float*     gng  = (const float*)d_in[12];
    const float*     gnb  = (const float*)d_in[13];
    const long long* eidx = (const long long*)d_in[14];

    char* ws = (char*)d_ws;
    bf16_t* we1t = (bf16_t*)(ws);
    bf16_t* we2t = (bf16_t*)(ws + 98304);
    bf16_t* wn1t = (bf16_t*)(ws + 131072);
    bf16_t* wn2t = (bf16_t*)(ws + 196608);
    float*  agg  = (float*)(ws + 229376);

    float* out_x = (float*)d_out;
    float* out_e = out_x + (size_t)N_TOT * D_DIM;

    hipMemsetAsync(agg, 0, (size_t)N_TOT * D_DIM * sizeof(float), stream);

    const int prep_elems = 384 * 128 + 128 * 128 + 256 * 128 + 128 * 128;  // 114688
    prep_weights<<<(prep_elems + 255) / 256, 256, 0, stream>>>(
        We1, We2, Wn1, Wn2, we1t, we2t, wn1t, wn2t);

    edge_kernel<<<(E_TOT + 127) / 128, 256, 0, stream>>>(
        x, ea, be1, be2, geg, geb, eidx, we1t, we2t, out_e, agg);

    node_kernel<<<(N_TOT + 127) / 128, 256, 0, stream>>>(
        x, agg, bn1, bn2, gng, gnb, wn1t, wn2t, out_x);
}
